// GNNStack_40888088657949
// MI455X (gfx1250) — matrix operand order repaired and checked by reading, not compiled, not run on hardware
//
#include <hip/hip_runtime.h>
#include <hip/hip_bf16.h>
#include <stdint.h>

// ---------------------------------------------------------------------------
// CDNA5 / gfx1250 types
// ---------------------------------------------------------------------------
typedef __attribute__((ext_vector_type(16))) __bf16 v16bf;
typedef __attribute__((ext_vector_type(8)))  float  v8f;
typedef unsigned int u32x4 __attribute__((ext_vector_type(4)));
typedef int          i32x8 __attribute__((ext_vector_type(8)));
typedef int          i32x4 __attribute__((ext_vector_type(4)));

union BFrag {              // 16 bf16 = 32 bytes = 8 VGPRs/lane
  v16bf v;
  uint4 q[2];
};

constexpr int kD  = 128;   // hidden / feature dim
constexpr int kT  = 60;    // sequence length
constexpr int kLP = 20;    // long-path GRUs
constexpr int kN  = 512;   // nodes
constexpr int kE  = 16384; // edges
constexpr int kOUT = 2;

// native bf16 conversion (gfx1250 has hardware bf16 cvt)
__device__ __forceinline__ unsigned short f2bf(float f) {
  __bf16 h = (__bf16)f;
  return __builtin_bit_cast(unsigned short, h);
}
__device__ __forceinline__ unsigned int pack2(float lo, float hi) {
  return (unsigned int)f2bf(lo) | ((unsigned int)f2bf(hi) << 16);
}
__device__ __forceinline__ float sigmoidf_(float x) {
  return 1.0f / (1.0f + __expf(-x));
}

// ---------------------------------------------------------------------------
// Tensor Data Mover: async 2-D tile load (rows x 128 f32) global -> LDS.
// D# packed per CDNA5 ISA 8.3/8.4: group0 = {count/type/addr}, group1 =
// {data_size, dims, tile dims, dim0 stride}; groups 2/3 zero (2-D tensor).
// ---------------------------------------------------------------------------
__device__ __forceinline__ void tdm_load_tile_f32(const float* gptr,
                                                  unsigned lds_off,
                                                  unsigned row_stride_elems,
                                                  unsigned rows) {
  unsigned long long ga = (unsigned long long)(uintptr_t)gptr;
  u32x4 g0;
  g0[0] = 1u;                                  // count=1, user mode
  g0[1] = lds_off;                             // lds_addr (bytes)
  g0[2] = (unsigned)(ga & 0xFFFFFFFFu);        // global_addr[31:0]
  g0[3] = (unsigned)(ga >> 32) | (2u << 30);   // global_addr[56:32] | type=2
  i32x8 g1;
  g1[0] = (int)(2u << 16);                     // wg_mask=0, data_size=2 (4B)
  g1[1] = (int)(128u << 16);                   // tensor_dim0[15:0]=128 @[63:48]
  g1[2] = (int)(rows << 16);                   // tensor_dim1[15:0]  @[95:80]
  g1[3] = (int)(128u << 16);                   // tile_dim0=128      @[127:112]
  g1[4] = (int)rows;                           // tile_dim1          @[143:128]
  g1[5] = (int)row_stride_elems;               // tensor_dim0_stride[31:0]
  g1[6] = 0;                                   // stride hi / dim1_stride lo
  g1[7] = 0;
  i32x4 z4 = {0, 0, 0, 0};
#if __clang_major__ >= 23
  i32x8 z8 = {0, 0, 0, 0, 0, 0, 0, 0};
  __builtin_amdgcn_tensor_load_to_lds(g0, g1, z4, z4, z8, 0);
#else
  __builtin_amdgcn_tensor_load_to_lds(g0, g1, z4, z4, 0);
#endif
}

// ---------------------------------------------------------------------------
// fp32 -> bf16 weight repack
// ---------------------------------------------------------------------------
__global__ void k_f2bf(const float* __restrict__ src,
                       unsigned short* __restrict__ dst, int n) {
  int i = blockIdx.x * blockDim.x + threadIdx.x;
  if (i < n) dst[i] = f2bf(src[i]);
}

// ---------------------------------------------------------------------------
// Fused GRU: 21 channels (20 lp + 1 ns), 16-node tiles, T=60 steps.
// Block = 256 threads = 8 waves. Wave w owns hidden cols [16w,16w+16) of all
// three gates, so WMMA C-layout lines up with the elementwise GRU update.
// x_t tiles are double-buffered in LDS via the Tensor Data Mover (wave 0
// issues tensor_load_to_lds for t+1 while everyone computes step t).
// ---------------------------------------------------------------------------
__global__ __launch_bounds__(256)
void k_gru(const float* __restrict__ x_lp, const float* __restrict__ x_ns,
           const unsigned short* __restrict__ WihLp,
           const unsigned short* __restrict__ WhhLp,
           const float* __restrict__ bihLp, const float* __restrict__ bhhLp,
           const unsigned short* __restrict__ WihNs,
           const unsigned short* __restrict__ WhhNs,
           const float* __restrict__ bihNs, const float* __restrict__ bhhNs,
           float* __restrict__ h_all) {
  __shared__ alignas(16) float          sXF[2][16 * kD];  // TDM landing (f32)
  __shared__ alignas(16) unsigned short sX[16 * kD];      // x_t tile, bf16
  __shared__ alignas(16) unsigned short sH[16 * kD];      // h tile,  bf16

  const int c    = blockIdx.y;            // channel: 0..19 lp, 20 = ns
  const int m0   = blockIdx.x * 16;       // node-tile base
  const int tid  = threadIdx.x;
  const int wv   = tid >> 5;
  const int lane = tid & 31;
  const int hf   = lane >> 4;             // lane half (0/1)
  const int l16  = lane & 15;

  const unsigned short* Wih;
  const unsigned short* Whh;
  const float* bih;
  const float* bhh;
  if (c < kLP) {
    Wih = WihLp + (size_t)c * 3 * kD * kD;
    Whh = WhhLp + (size_t)c * 3 * kD * kD;
    bih = bihLp + c * 3 * kD;
    bhh = bhhLp + c * 3 * kD;
  } else {
    Wih = WihNs; Whh = WhhNs; bih = bihNs; bhh = bhhNs;
  }

  // per-wave gate bias (column-dependent only)
  float bi[3], bh[3];
#pragma unroll
  for (int g = 0; g < 3; ++g) {
    int col = g * kD + wv * 16 + l16;
    bi[g] = bih[col];
    bh[g] = bhh[col];
  }

  // h accumulator in C-layout: VGPR i, row M = i + 8*hf, col = 16*wv + l16
  v8f hC;
#pragma unroll
  for (int i = 0; i < 8; ++i) hC[i] = 0.0f;

  // zero initial hidden state in LDS
#pragma unroll
  for (int i = 0; i < 8; ++i) sH[tid * 8 + i] = 0;

  // uniform tile geometry for the TDM
  const unsigned row_stride = (c < kLP) ? (unsigned)(kLP * kT * kD)
                                        : (unsigned)(kT * kD);
  const unsigned ldsA = (unsigned)(uintptr_t)(void*)&sXF[0][0];
  const unsigned ldsB = (unsigned)(uintptr_t)(void*)&sXF[1][0];

  // preload tile t=0
  if (wv == 0) {
    const float* p0 = (c < kLP)
        ? (x_lp + (((size_t)m0 * kLP + c) * kT + 0) * kD)
        : (x_ns + ((size_t)m0 * kT + 0) * kD);
    tdm_load_tile_f32(p0, ldsA, row_stride, 16u);
  }

  for (int t = 0; t < kT; ++t) {
    // ---- tile t has landed; publish to the workgroup --------------------
    if (wv == 0) __builtin_amdgcn_s_wait_tensorcnt(0);
    __syncthreads();

    // ---- kick off async load of tile t+1 into the other buffer ---------
    if (wv == 0 && t + 1 < kT) {
      const float* pn = (c < kLP)
          ? (x_lp + (((size_t)m0 * kLP + c) * kT + (t + 1)) * kD)
          : (x_ns + ((size_t)m0 * kT + (t + 1)) * kD);
      tdm_load_tile_f32(pn, (t & 1) ? ldsA : ldsB, row_stride, 16u);
    }

    // ---- convert f32 tile -> bf16 sX (cooperative) ----------------------
    {
      int m  = tid >> 4;
      int d0 = (tid & 15) << 3;
      const float* xp = &sXF[t & 1][m * kD + d0];
      float4 a  = ((const float4*)xp)[0];
      float4 b2 = ((const float4*)xp)[1];
      unsigned int* sp = (unsigned int*)&sX[m * kD + d0];
      sp[0] = pack2(a.x, a.y);   sp[1] = pack2(a.z, a.w);
      sp[2] = pack2(b2.x, b2.y); sp[3] = pack2(b2.z, b2.w);
    }
    __syncthreads();

    // ---- build A fragments (16x32 bf16, ISA layout) --------------------
    BFrag aX[4], aH[4];
#pragma unroll
    for (int ks = 0; ks < 4; ++ks) {
      int kb = ks * 32 + 8 * hf;  // run1 start; run2 = +16
      const uint4* px = (const uint4*)&sX[l16 * kD + kb];
      aX[ks].q[0] = px[0]; aX[ks].q[1] = px[2];
      const uint4* ph = (const uint4*)&sH[l16 * kD + kb];
      aH[ks].q[0] = ph[0]; aH[ks].q[1] = ph[2];
    }

    // ---- gi = x@Wih^T + bih ; gh = h@Whh^T + bhh  (WMMA bf16) ----------
    v8f gi[3], gh[3];
#pragma unroll
    for (int g = 0; g < 3; ++g) {
#pragma unroll
      for (int i = 0; i < 8; ++i) { gi[g][i] = bi[g]; gh[g][i] = bh[g]; }
      const int nrow = g * kD + wv * 16 + l16;   // weight row (bias col)
#pragma unroll
      for (int ks = 0; ks < 4; ++ks) {
        BFrag bW, bV;
        const uint4* pw =
            (const uint4*)&Wih[(size_t)nrow * kD + ks * 32 + 8 * hf];
        bW.q[0] = pw[0]; bW.q[1] = pw[2];
        gi[g] = __builtin_amdgcn_wmma_f32_16x16x32_bf16(
            false, aX[ks].v, false, bW.v, (short)0, gi[g], false, false);
        const uint4* pv =
            (const uint4*)&Whh[(size_t)nrow * kD + ks * 32 + 8 * hf];
        bV.q[0] = pv[0]; bV.q[1] = pv[2];
        gh[g] = __builtin_amdgcn_wmma_f32_16x16x32_bf16(
            false, aH[ks].v, false, bV.v, (short)0, gh[g], false, false);
      }
    }
    __syncthreads();   // all reads of sH/sX done before overwrite

    // ---- elementwise GRU update (layouts line up per-lane) -------------
#pragma unroll
    for (int i = 0; i < 8; ++i) {
      float r  = sigmoidf_(gi[0][i] + gh[0][i]);
      float z  = sigmoidf_(gi[1][i] + gh[1][i]);
      float nn = tanhf(gi[2][i] + r * gh[2][i]);
      hC[i] = (1.0f - z) * nn + z * hC[i];
    }
    // write h_new back to LDS (bf16) in C layout
    {
      int col = wv * 16 + l16;
#pragma unroll
      for (int i = 0; i < 8; ++i) sH[(i + 8 * hf) * kD + col] = f2bf(hC[i]);
    }
    __syncthreads();
  }

  // ---- final hidden state -> h_all[c, node, :] ------------------------
  {
    int col = wv * 16 + l16;
#pragma unroll
    for (int i = 0; i < 8; ++i) {
      int Mr = i + 8 * hf;
      h_all[((size_t)c * kN + (m0 + Mr)) * kD + col] = hC[i];
    }
  }
}

// ---------------------------------------------------------------------------
// Generic WMMA bf16 GEMM: C[M,Nn] = A[M,K] @ W[Nn,K]^T + bias
// block = 128 threads (4 waves); block tile 16 x 64; K multiple of 32.
// ---------------------------------------------------------------------------
__global__ __launch_bounds__(128)
void k_gemm(const float* __restrict__ A, const float* __restrict__ W,
            const float* __restrict__ bias, float* __restrict__ C,
            int M, int K, int Nn) {
  const int tid  = threadIdx.x;
  const int wv   = tid >> 5;
  const int lane = tid & 31;
  const int hf   = lane >> 4;
  const int l16  = lane & 15;
  const int m0   = blockIdx.x * 16;
  const int n    = blockIdx.y * 64 + wv * 16 + l16;
  const int nc   = (n < Nn) ? n : (Nn - 1);

  v8f acc;
#pragma unroll
  for (int i = 0; i < 8; ++i) acc[i] = 0.0f;

  const float* arow = A + (size_t)(m0 + l16) * K;
  const float* wrow = W + (size_t)nc * K;

  for (int kb = 0; kb < K; kb += 32) {
    BFrag a, b;
    const float* ap = arow + kb + 8 * hf;   // two runs of 8
    a.q[0].x = pack2(ap[0],  ap[1]);  a.q[0].y = pack2(ap[2],  ap[3]);
    a.q[0].z = pack2(ap[4],  ap[5]);  a.q[0].w = pack2(ap[6],  ap[7]);
    a.q[1].x = pack2(ap[16], ap[17]); a.q[1].y = pack2(ap[18], ap[19]);
    a.q[1].z = pack2(ap[20], ap[21]); a.q[1].w = pack2(ap[22], ap[23]);
    const float* wp = wrow + kb + 8 * hf;   // two runs of 8
    b.q[0].x = pack2(wp[0],  wp[1]);  b.q[0].y = pack2(wp[2],  wp[3]);
    b.q[0].z = pack2(wp[4],  wp[5]);  b.q[0].w = pack2(wp[6],  wp[7]);
    b.q[1].x = pack2(wp[16], wp[17]); b.q[1].y = pack2(wp[18], wp[19]);
    b.q[1].z = pack2(wp[20], wp[21]); b.q[1].w = pack2(wp[22], wp[23]);
    acc = __builtin_amdgcn_wmma_f32_16x16x32_bf16(
        false, a.v, false, b.v, (short)0, acc, false, false);
  }

  if (n < Nn) {
    float bv = bias ? bias[n] : 0.0f;
#pragma unroll
    for (int i = 0; i < 8; ++i) {
      int Mr = m0 + i + 8 * hf;
      C[(size_t)Mr * Nn + n] = acc[i] + bv;
    }
  }
}

// ---------------------------------------------------------------------------
// Small helper kernels
// ---------------------------------------------------------------------------
__global__ void k_fill_f(float* __restrict__ p, float v, int n) {
  int i = blockIdx.x * blockDim.x + threadIdx.x;
  if (i < n) p[i] = v;
}
__global__ void k_fill_u32(unsigned int* __restrict__ p, unsigned int v, int n) {
  int i = blockIdx.x * blockDim.x + threadIdx.x;
  if (i < n) p[i] = v;
}
// cat5: [x_pdt; x_ref; x_def; lp_feat; h_ns] flat -> xcat (reshape is a view)
__global__ void k_cat5(const float* __restrict__ p0, const float* __restrict__ p1,
                       const float* __restrict__ p2, const float* __restrict__ p3,
                       const float* __restrict__ p4, float* __restrict__ out) {
  int i = blockIdx.x * blockDim.x + threadIdx.x;
  if (i >= 5 * kN * kD) return;
  int b = i >> 16;           // 65536 = kN*kD per source
  int r = i & 65535;
  const float* s = (b == 0) ? p0 : (b == 1) ? p1 : (b == 2) ? p2
                 : (b == 3) ? p3 : p4;
  out[i] = s[r];
}
__global__ void k_deg(const int* __restrict__ ei, unsigned int* __restrict__ deg) {
  int e = blockIdx.x * blockDim.x + threadIdx.x;
  if (e < kE) atomicAdd(&deg[ei[kE + e]], 1u);
}
__global__ void k_dinv(const unsigned int* __restrict__ deg,
                       float* __restrict__ dinv) {
  int n = blockIdx.x * blockDim.x + threadIdx.x;
  if (n < kN) {
    float d = (float)(deg[n] + 1u);   // +1 self-loop
    dinv[n] = (d > 0.0f) ? rsqrtf(d) : 0.0f;
  }
}
__global__ void k_scatter(const float* __restrict__ xw, const int* __restrict__ ei,
                          const float* __restrict__ dinv, float* __restrict__ out) {
  int i = blockIdx.x * blockDim.x + threadIdx.x;   // over E * D
  if (i >= kE * kD) return;
  int e = i >> 7, d = i & 127;
  int s = ei[e], t = ei[kE + e];
  float v = xw[(size_t)s * kD + d] * dinv[s] * dinv[t];
  atomicAdd(&out[(size_t)t * kD + d], v);
}
__global__ void k_finalize(const float* __restrict__ xw,
                           const float* __restrict__ dinv,
                           const float* __restrict__ b, float* __restrict__ out) {
  int i = blockIdx.x * blockDim.x + threadIdx.x;
  if (i >= kN * kD) return;
  int n = i >> 7, d = i & 127;
  out[i] += xw[i] * dinv[n] * dinv[n] + b[d];
}
// relu then layernorm; one wave per node (wave32 shfl reduction)
__global__ __launch_bounds__(128)
void k_relu_ln(const float* __restrict__ x, const float* __restrict__ g,
               const float* __restrict__ b, float* __restrict__ y) {
  int wv = threadIdx.x >> 5, lane = threadIdx.x & 31;
  int n = blockIdx.x * 4 + wv;
  const float* xp = x + (size_t)n * kD;
  float v[4];
  float s = 0.0f;
#pragma unroll
  for (int i = 0; i < 4; ++i) {
    float t = xp[lane * 4 + i];
    t = t > 0.0f ? t : 0.0f;
    v[i] = t; s += t;
  }
  for (int o = 16; o > 0; o >>= 1) s += __shfl_xor(s, o, 32);
  float mean = s * (1.0f / kD);
  float vs = 0.0f;
#pragma unroll
  for (int i = 0; i < 4; ++i) { float d = v[i] - mean; vs += d * d; }
  for (int o = 16; o > 0; o >>= 1) vs += __shfl_xor(vs, o, 32);
  float inv = rsqrtf(vs * (1.0f / kD) + 1e-5f);
#pragma unroll
  for (int i = 0; i < 4; ++i) {
    int d = lane * 4 + i;
    y[(size_t)n * kD + d] = (v[i] - mean) * inv * g[d] + b[d];
  }
}
__global__ void k_emb_relu(const float* __restrict__ y, float* __restrict__ emb,
                           float* __restrict__ x3) {
  int i = blockIdx.x * blockDim.x + threadIdx.x;
  if (i < kN * kD) {
    float v = y[i];
    emb[i] = v;
    x3[i] = v > 0.0f ? v : 0.0f;
  }
}
__global__ void k_lsm(const float* __restrict__ z, float* __restrict__ out) {
  int n = blockIdx.x * blockDim.x + threadIdx.x;
  if (n < kN) {
    float a = z[n * 2], b = z[n * 2 + 1];
    float m = fmaxf(a, b);
    float l = m + __logf(__expf(a - m) + __expf(b - m));
    out[n * 2] = a - l;
    out[n * 2 + 1] = b - l;
  }
}

// ---------------------------------------------------------------------------
// Host launcher
// ---------------------------------------------------------------------------
extern "C" void kernel_launch(void* const* d_in, const int* in_sizes, int n_in,
                              void* d_out, int out_size, void* d_ws, size_t ws_size,
                              hipStream_t stream) {
  (void)in_sizes; (void)n_in; (void)out_size; (void)ws_size;
  const float* x_lp   = (const float*)d_in[0];
  const float* x_ns   = (const float*)d_in[1];
  const float* x_ref  = (const float*)d_in[2];
  const float* x_def  = (const float*)d_in[3];
  const float* x_pdt  = (const float*)d_in[4];
  const int*   eidx   = (const int*)d_in[5];
  const float* Wih_lp = (const float*)d_in[6];
  const float* Whh_lp = (const float*)d_in[7];
  const float* bih_lp = (const float*)d_in[8];
  const float* bhh_lp = (const float*)d_in[9];
  const float* lpfc_W = (const float*)d_in[10];
  const float* lpfc_b = (const float*)d_in[11];
  const float* Wih_ns = (const float*)d_in[12];
  const float* Whh_ns = (const float*)d_in[13];
  const float* bih_ns = (const float*)d_in[14];
  const float* bhh_ns = (const float*)d_in[15];
  const float* alfc_W = (const float*)d_in[16];
  const float* alfc_b = (const float*)d_in[17];
  const float* convW[3] = {(const float*)d_in[18], (const float*)d_in[20],
                           (const float*)d_in[22]};
  const float* convB[3] = {(const float*)d_in[19], (const float*)d_in[21],
                           (const float*)d_in[23]};
  const float* lng[2] = {(const float*)d_in[24], (const float*)d_in[26]};
  const float* lnb[2] = {(const float*)d_in[25], (const float*)d_in[27]};
  const float* mp1_W = (const float*)d_in[28];
  const float* mp1_b = (const float*)d_in[29];
  const float* mp2_W = (const float*)d_in[30];
  const float* mp2_b = (const float*)d_in[31];

  // workspace carve-up (256B aligned slots)
  size_t off = 0;
  auto carve = [&](size_t bytes) -> char* {
    char* p = (char*)d_ws + off;
    off += (bytes + 255) & ~(size_t)255;
    return p;
  };
  unsigned short* WihLp_bf = (unsigned short*)carve((size_t)kLP * 3 * kD * kD * 2);
  unsigned short* WhhLp_bf = (unsigned short*)carve((size_t)kLP * 3 * kD * kD * 2);
  unsigned short* WihNs_bf = (unsigned short*)carve((size_t)3 * kD * kD * 2);
  unsigned short* WhhNs_bf = (unsigned short*)carve((size_t)3 * kD * kD * 2);
  float* h_all   = (float*)carve((size_t)(kLP + 1) * kN * kD * 4);
  float* lp_feat = (float*)carve((size_t)kN * kD * 4);
  float* xcat5   = (float*)carve((size_t)kN * 5 * kD * 4);
  float* bufX    = (float*)carve((size_t)kN * kD * 4);
  float* bufX2   = (float*)carve((size_t)kN * kD * 4);
  float* bufXW   = (float*)carve((size_t)kN * kD * 4);
  float* bufY    = (float*)carve((size_t)kN * kD * 4);
  float* z1      = (float*)carve((size_t)kN * kD * 4);
  float* z2      = (float*)carve((size_t)kN * kOUT * 4);
  unsigned int* deg = (unsigned int*)carve((size_t)kN * 4);
  float* dinv    = (float*)carve((size_t)kN * 4);

  const int B = 256;
  // 1) weights -> bf16
  int nw = kLP * 3 * kD * kD;
  k_f2bf<<<(nw + B - 1) / B, B, 0, stream>>>(Wih_lp, WihLp_bf, nw);
  k_f2bf<<<(nw + B - 1) / B, B, 0, stream>>>(Whh_lp, WhhLp_bf, nw);
  int nw2 = 3 * kD * kD;
  k_f2bf<<<(nw2 + B - 1) / B, B, 0, stream>>>(Wih_ns, WihNs_bf, nw2);
  k_f2bf<<<(nw2 + B - 1) / B, B, 0, stream>>>(Whh_ns, WhhNs_bf, nw2);

  // 2) all 21 GRUs (WMMA bf16 + TDM double-buffered x tiles)
  k_gru<<<dim3(kN / 16, kLP + 1), 256, 0, stream>>>(
      x_lp, x_ns, WihLp_bf, WhhLp_bf, bih_lp, bhh_lp,
      WihNs_bf, WhhNs_bf, bih_ns, bhh_ns, h_all);

  // 3) lp_fc : h_lp.reshape(N, LP*D) is a view of h_all[0:20]
  k_gemm<<<dim3(kN / 16, 2), 128, 0, stream>>>(h_all, lpfc_W, lpfc_b, lp_feat,
                                               kN, kLP * kD, kD);
  // 4) cat5 (concat+reshape is a view -> flat gather) then all_fc
  k_cat5<<<(5 * kN * kD + B - 1) / B, B, 0, stream>>>(
      x_pdt, x_ref, x_def, lp_feat, h_all + (size_t)kLP * kN * kD, xcat5);
  k_gemm<<<dim3(kN / 16, 2), 128, 0, stream>>>(xcat5, alfc_W, alfc_b, bufX,
                                               kN, 5 * kD, kD);

  // 5) GCN degree / norm
  k_fill_u32<<<(kN + B - 1) / B, B, 0, stream>>>(deg, 0u, kN);
  k_deg<<<(kE + B - 1) / B, B, 0, stream>>>(eidx, deg);
  k_dinv<<<(kN + B - 1) / B, B, 0, stream>>>(deg, dinv);

  // 6) 3 GCN layers
  float* xin = bufX;
  float* xout = bufX2;
  for (int l = 0; l < 3; ++l) {
    k_gemm<<<dim3(kN / 16, 2), 128, 0, stream>>>(xin, convW[l], nullptr, bufXW,
                                                 kN, kD, kD);
    k_fill_f<<<(kN * kD + B - 1) / B, B, 0, stream>>>(bufY, 0.0f, kN * kD);
    k_scatter<<<(kE * kD + B - 1) / B, B, 0, stream>>>(bufXW, eidx, dinv, bufY);
    k_finalize<<<(kN * kD + B - 1) / B, B, 0, stream>>>(bufXW, dinv, convB[l], bufY);
    if (l < 2) {
      k_relu_ln<<<kN / 4, 128, 0, stream>>>(bufY, lng[l], lnb[l], xout);
      xin = xout;
      xout = (xin == bufX2) ? bufX : bufX2;
    } else {
      // emb = conv2 output -> d_out[0 : N*D); relu -> xin for mp1
      k_emb_relu<<<(kN * kD + B - 1) / B, B, 0, stream>>>(bufY, (float*)d_out, bufX);
      xin = bufX;
    }
  }

  // 7) mp1, mp2, log_softmax
  k_gemm<<<dim3(kN / 16, 2), 128, 0, stream>>>(xin, mp1_W, mp1_b, z1, kN, kD, kD);
  k_gemm<<<dim3(kN / 16, 1), 128, 0, stream>>>(z1, mp2_W, mp2_b, z2, kN, kD, kOUT);
  k_lsm<<<(kN + B - 1) / B, B, 0, stream>>>(z2, (float*)d_out + (size_t)kN * kD);
}
